// Qwen2Model_9474697855270
// MI455X (gfx1250) — matrix operand order repaired and checked by reading, not compiled, not run on hardware
//
#include <hip/hip_runtime.h>
#include <hip/hip_bf16.h>

// ---------------- model constants ----------------
#define BB   2
#define SS   2048
#define DD   1536
#define HH   12
#define KVH  2
#define DH   128
#define II   8960
#define LL   4
#define MM   (BB*SS)          // 4096 token rows
#define KVD  (KVH*DH)         // 256
#define EPS  1e-6f

typedef __attribute__((ext_vector_type(16))) __bf16 v16bf;
typedef __attribute__((ext_vector_type(8)))  float  v8f;

// ---------------- WMMA fragment loaders ----------------
// A fragment: 16x32 bf16, row-major source with element stride `stride`.
// lanes 0-15: row=lane, K 0..7 (v0-3) and 16..23 (v4-7)
// lanes16-31: row=lane-15... row=lane-16, K 8..15 and 24..31
__device__ __forceinline__ v16bf load_frag_a(const __bf16* base, int stride, int lane) {
    int row = lane & 15;
    int ko  = (lane < 16) ? 0 : 8;
    const __bf16* p = base + (size_t)row * stride + ko;
    union { uint4 u[2]; v16bf v; } U;
    U.u[0] = *reinterpret_cast<const uint4*>(p);
    U.u[1] = *reinterpret_cast<const uint4*>(p + 16);
    return U.v;
}

// B fragment: 32(K)x16(N) bf16 from a transposed [N][K] source (contiguous K).
// lanes 0-15: col=lane, K 0..7,16..23 ; lanes16-31: col=lane-16, K 8..15,24..31
__device__ __forceinline__ v16bf load_frag_b(const __bf16* base, int stride, int lane) {
    int col = lane & 15;
    int ko  = (lane < 16) ? 0 : 8;
    const __bf16* p = base + (size_t)col * stride + ko;
    union { uint4 u[2]; v16bf v; } U;
    U.u[0] = *reinterpret_cast<const uint4*>(p);
    U.u[1] = *reinterpret_cast<const uint4*>(p + 16);
    return U.v;
}

// B fragment from per-lane prepped pointer (16B at p, 16B at p+16 elems)
__device__ __forceinline__ v16bf fragB_at(const __bf16* p) {
    union { uint4 u[2]; v16bf v; } U;
    U.u[0] = *reinterpret_cast<const uint4*>(p);
    U.u[1] = *reinterpret_cast<const uint4*>(p + 16);
    return U.v;
}

// A fragment from per-lane prepped pointer (16B at p, 16B at p+16 elems)
__device__ __forceinline__ v16bf fragA_at(const __bf16* p) {
    union { uint4 u[2]; v16bf v; } U;
    U.u[0] = *reinterpret_cast<const uint4*>(p);
    U.u[1] = *reinterpret_cast<const uint4*>(p + 16);
    return U.v;
}

__device__ __forceinline__ v8f wmma_bf16(v16bf a, v16bf b, v8f c) {
    return __builtin_amdgcn_wmma_f32_16x16x32_bf16(false, a, false, b, (short)0, c, false, false);
}

// CDNA5 async global->LDS copy (ASYNCcnt path), 16 bytes per lane.
__device__ __forceinline__ void async_ld_b128(unsigned lds_off, const __bf16* g) {
    asm volatile("global_load_async_to_lds_b128 %0, %1, off"
                 :: "v"(lds_off), "v"(g) : "memory");
}
__device__ __forceinline__ void wait_asynccnt0() {
    asm volatile("s_wait_asynccnt 0x0" ::: "memory");
}

// ---------------- embedding gather ----------------
__global__ __launch_bounds__(256) void k_embed(const int* __restrict__ ids,
                                               const float* __restrict__ emb,
                                               float* __restrict__ h) {
    int t = blockIdx.x;
    int id = ids[t];
    const float* src = emb + (size_t)id * DD;
    float* dst = h + (size_t)t * DD;
    for (int i = threadIdx.x; i < DD; i += 256) dst[i] = src[i];
}

// ---------------- weight convert + transpose: W[K][N] f32 -> WT[N][K] bf16 ----
__global__ __launch_bounds__(256) void k_cvt_wT(const float* __restrict__ W,
                                                __bf16* __restrict__ WT,
                                                int K, int N) {
    __shared__ float tile[32][33];
    int nb = blockIdx.x * 32, kb = blockIdx.y * 32;
    int tx = threadIdx.x & 31, ty = threadIdx.x >> 5;  // 32 x 8
    for (int r = ty; r < 32; r += 8) {
        int k = kb + r, n = nb + tx;
        tile[r][tx] = (k < K && n < N) ? W[(size_t)k * N + n] : 0.f;
    }
    __syncthreads();
    for (int r = ty; r < 32; r += 8) {
        int n = nb + r, k = kb + tx;
        if (n < N && k < K) WT[(size_t)n * K + k] = (__bf16)tile[tx][r];
    }
}

// ---------------- RMSNorm: fp32 in -> bf16 (and/or fp32) out ----------------
__global__ __launch_bounds__(256) void k_rmsnorm(const float* __restrict__ x,
                                                 const float* __restrict__ w,
                                                 __bf16* __restrict__ outB,
                                                 float* __restrict__ outF) {
    int row = blockIdx.x, tid = threadIdx.x;
    const float* xr = x + (size_t)row * DD;
    float ss = 0.f;
    for (int i = tid; i < DD; i += 256) { float v = xr[i]; ss += v * v; }
    for (int o = 16; o > 0; o >>= 1) ss += __shfl_xor(ss, o, 32);
    __shared__ float red[8];
    if ((tid & 31) == 0) red[tid >> 5] = ss;
    __syncthreads();
    float tot = 0.f;
    for (int i = 0; i < 8; i++) tot += red[i];
    float rinv = rsqrtf(tot / (float)DD + EPS);
    for (int i = tid; i < DD; i += 256) {
        float v = xr[i] * rinv * w[i];
        if (outB) outB[(size_t)row * DD + i] = (__bf16)v;
        if (outF) outF[(size_t)row * DD + i] = v;
    }
}

// ---------------- WMMA GEMM with async-LDS A staging + pipelined B -----------
// C[M][N] = A[M][K](bf16) x BT[N][K](bf16)
// Block: 256 threads = 8 waves, tile 32(M) x 256(N).
// A tile (32 x 32 per K-step) staged in LDS via global_load_async_to_lds_b128,
// double buffered; every wave reads A fragments from LDS (4x reuse).
// flags: 1=add bias, 2=residual accumulate into Cf, 4=bf16 output to Cb
__global__ __launch_bounds__(256) void k_gemm(const __bf16* __restrict__ A,
                                              const __bf16* __restrict__ BT,
                                              const float* __restrict__ bias,
                                              float* __restrict__ Cf,
                                              __bf16* __restrict__ Cb,
                                              int M, int N, int K, int flags) {
    __shared__ __attribute__((aligned(16))) __bf16 atile[2][32 * 32];   // 2 x 2KB
    int tid  = threadIdx.x;
    int lane = tid & 31;
    int wave = tid >> 5;                    // 8 waves
    int mblk  = blockIdx.y * 32;
    int mtile = (wave & 1) * 16;            // within block
    int nbase = blockIdx.x * 256 + (wave >> 1) * 64;

    // ---- async A-stager: 128 threads move 32 rows x 64B per K-step ----
    bool loader = tid < 128;
    int lrow = tid >> 2, lseg = tid & 3;                       // row 0..31, 16B seg 0..3
    const __bf16* Ag = A + (size_t)(mblk + lrow) * K + lseg * 8;
    unsigned lds0 = (unsigned)(uintptr_t)(&atile[0][0]) + (unsigned)(lrow * 64 + lseg * 16);
    unsigned lds1 = lds0 + 2048;

    // ---- per-lane prepped pointers ----
    // LDS A fragment pointer for this wave's 16-row subtile
    const __bf16* aP0 = &atile[0][0] + mtile * 32 + (lane & 15) * 32 + ((lane < 16) ? 0 : 8);
    const __bf16* aP1 = aP0 + 1024;
    // global B fragment pointers (advance by 32 elems per K-step)
    size_t lB = (size_t)(lane & 15) * K + ((lane < 16) ? 0 : 8);
    const __bf16* b0p = BT + (size_t)(nbase +  0) * K + lB;
    const __bf16* b1p = BT + (size_t)(nbase + 16) * K + lB;
    const __bf16* b2p = BT + (size_t)(nbase + 32) * K + lB;
    const __bf16* b3p = BT + (size_t)(nbase + 48) * K + lB;

    v8f acc0 = {}, acc1 = {}, acc2 = {}, acc3 = {};
    v16bf bA0, bA1, bA2, bA3, bB0, bB1, bB2, bB3;

    // prologue: stage K-step 0 into buf0, preload B regs for K-step 0
    if (loader) async_ld_b128(lds0, Ag);
    bA0 = fragB_at(b0p); bA1 = fragB_at(b1p);
    bA2 = fragB_at(b2p); bA3 = fragB_at(b3p);

    int k = 0;
    for (;;) {
        // ---- even step: consume buf0 / bA, prefetch buf1 / bB for k+32 ----
        wait_asynccnt0();
        __syncthreads();
        {
            bool more = (k + 32 < K);
            if (more) {
                if (loader) async_ld_b128(lds1, Ag + (k + 32));
                __builtin_prefetch(b0p + (k + 128), 0, 1);
                bB0 = fragB_at(b0p + (k + 32)); bB1 = fragB_at(b1p + (k + 32));
                bB2 = fragB_at(b2p + (k + 32)); bB3 = fragB_at(b3p + (k + 32));
            }
            v16bf a = fragA_at(aP0);
            acc0 = wmma_bf16(a, bA0, acc0);
            acc1 = wmma_bf16(a, bA1, acc1);
            acc2 = wmma_bf16(a, bA2, acc2);
            acc3 = wmma_bf16(a, bA3, acc3);
            k += 32;
            if (!more) break;
        }
        // ---- odd step: consume buf1 / bB, prefetch buf0 / bA for k+32 ----
        wait_asynccnt0();
        __syncthreads();
        {
            bool more = (k + 32 < K);
            if (more) {
                if (loader) async_ld_b128(lds0, Ag + (k + 32));
                __builtin_prefetch(b2p + (k + 128), 0, 1);
                bA0 = fragB_at(b0p + (k + 32)); bA1 = fragB_at(b1p + (k + 32));
                bA2 = fragB_at(b2p + (k + 32)); bA3 = fragB_at(b3p + (k + 32));
            }
            v16bf a = fragA_at(aP1);
            acc0 = wmma_bf16(a, bB0, acc0);
            acc1 = wmma_bf16(a, bB1, acc1);
            acc2 = wmma_bf16(a, bB2, acc2);
            acc3 = wmma_bf16(a, bB3, acc3);
            k += 32;
            if (!more) break;
        }
    }

    int m_off = (lane < 16) ? 0 : 8;
    int ncol  = lane & 15;
    float bj0 = 0.f, bj1 = 0.f, bj2 = 0.f, bj3 = 0.f;
    if (flags & 1) {
        bj0 = bias[nbase + 0  + ncol];
        bj1 = bias[nbase + 16 + ncol];
        bj2 = bias[nbase + 32 + ncol];
        bj3 = bias[nbase + 48 + ncol];
    }
    for (int r = 0; r < 8; r++) {
        int m = mblk + mtile + m_off + r;
        float v0 = acc0[r] + bj0, v1 = acc1[r] + bj1;
        float v2 = acc2[r] + bj2, v3 = acc3[r] + bj3;
        size_t idx = (size_t)m * N + nbase + ncol;
        if (flags & 2) {
            Cf[idx] += v0; Cf[idx + 16] += v1; Cf[idx + 32] += v2; Cf[idx + 48] += v3;
        } else if (flags & 4) {
            Cb[idx] = (__bf16)v0; Cb[idx + 16] = (__bf16)v1;
            Cb[idx + 32] = (__bf16)v2; Cb[idx + 48] = (__bf16)v3;
        } else {
            Cf[idx] = v0; Cf[idx + 16] = v1; Cf[idx + 32] = v2; Cf[idx + 48] = v3;
        }
    }
}

// ---------------- RoPE: fp32 qf/kf -> bf16 head-major qb/kb ----------------
__global__ __launch_bounds__(64) void k_rope(const float* __restrict__ qf,
                                             const float* __restrict__ kf,
                                             __bf16* __restrict__ qb,
                                             __bf16* __restrict__ kb) {
    int t = blockIdx.x;                 // token 0..MM-1
    int b = t / SS, s = t % SS;
    int hh = blockIdx.y;                // 0..HH-1 = q heads, HH..HH+KVH-1 = k heads
    int j = threadIdx.x;                // 0..63 rotation pairs
    float inv = __powf(1.0e6f, -(float)j / 64.f);
    float ang = (float)s * inv;
    float c = __cosf(ang), sn = __sinf(ang);
    if (hh < HH) {
        const float* src = qf + (size_t)t * DD + hh * DH;
        __bf16* dst = qb + ((size_t)(b * HH + hh) * SS + s) * DH;
        float x1 = src[j], x2 = src[j + 64];
        dst[j]      = (__bf16)(x1 * c - x2 * sn);
        dst[j + 64] = (__bf16)(x2 * c + x1 * sn);
    } else {
        int kv = hh - HH;
        const float* src = kf + (size_t)t * KVD + kv * DH;
        __bf16* dst = kb + ((size_t)(b * KVH + kv) * SS + s) * DH;
        float x1 = src[j], x2 = src[j + 64];
        dst[j]      = (__bf16)(x1 * c - x2 * sn);
        dst[j + 64] = (__bf16)(x2 * c + x1 * sn);
    }
}

// ---------------- V reorder: fp32 [t][kv*DH] -> bf16 transposed [b][kv][d][s] ---
__global__ __launch_bounds__(256) void k_vT(const float* __restrict__ vf,
                                            __bf16* __restrict__ vt) {
    int t = blockIdx.x, i = threadIdx.x;        // i in 0..255
    int b = t / SS, s = t % SS;
    int kv = i >> 7, d = i & 127;
    vt[((size_t)(b * KVH + kv) * DH + d) * SS + s] = (__bf16)vf[(size_t)t * KVD + i];
}

// ---------------- flash attention: 1 wave per (b, h, 16-query tile) ----------
__global__ __launch_bounds__(32) void k_attn(const __bf16* __restrict__ qb,
                                             const __bf16* __restrict__ kb,
                                             const __bf16* __restrict__ vt,
                                             __bf16* __restrict__ aob) {
    __shared__ __attribute__((aligned(16))) __bf16 pls[16 * 32];
    const float scale = 0.088388347648318447f;  // DH^-0.5
    int lane = threadIdx.x;
    int qs = blockIdx.x * 16;
    int h  = blockIdx.y, b = blockIdx.z;
    int kvh = h / (HH / KVH);

    const __bf16* Q  = qb + ((size_t)(b * HH + h) * SS + qs) * DH;
    const __bf16* Kp = kb + (size_t)(b * KVH + kvh) * SS * DH;
    const __bf16* Vt = vt + (size_t)(b * KVH + kvh) * DH * SS;

    v16bf qa[4];
    for (int d = 0; d < 4; d++) qa[d] = load_frag_a(Q + d * 32, DH, lane);

    v8f accO[8]; for (int j = 0; j < 8; j++) accO[j] = (v8f){};
    float mi[8], li[8];
    for (int r = 0; r < 8; r++) { mi[r] = -1e30f; li[r] = 0.f; }

    int m_off = (lane < 16) ? 0 : 8;
    int nidx  = lane & 15;

    for (int ks = 0; ks < qs + 16; ks += 32) {
        v8f s0 = {}, s1 = {};
        for (int d = 0; d < 4; d++) {
            v16bf kb0 = load_frag_b(Kp + (size_t)ks * DH + d * 32, DH, lane);
            v16bf kb1 = load_frag_b(Kp + (size_t)(ks + 16) * DH + d * 32, DH, lane);
            s0 = wmma_bf16(qa[d], kb0, s0);
            s1 = wmma_bf16(qa[d], kb1, s1);
        }
        float alpha[8];
        for (int r = 0; r < 8; r++) {
            int qrow = qs + m_off + r;
            float v0 = s0[r] * scale, v1 = s1[r] * scale;
            if (ks + nidx      > qrow) v0 = -1e30f;
            if (ks + 16 + nidx > qrow) v1 = -1e30f;
            float rm = fmaxf(v0, v1);
            for (int o = 8; o > 0; o >>= 1) rm = fmaxf(rm, __shfl_xor(rm, o, 32));
            float mnew = fmaxf(mi[r], rm);
            alpha[r] = __expf(mi[r] - mnew);
            mi[r] = mnew;
            float p0 = __expf(v0 - mnew), p1 = __expf(v1 - mnew);
            float rs = p0 + p1;
            for (int o = 8; o > 0; o >>= 1) rs += __shfl_xor(rs, o, 32);
            li[r] = li[r] * alpha[r] + rs;
            // stash P into LDS in row-major 16x32 for the A-fragment reload
            pls[(m_off + r) * 32 + nidx]      = (__bf16)p0;
            pls[(m_off + r) * 32 + 16 + nidx] = (__bf16)p1;
        }
        for (int j = 0; j < 8; j++)
            for (int r = 0; r < 8; r++) accO[j][r] *= alpha[r];
        asm volatile("s_wait_dscnt 0x0" ::: "memory");   // P stores visible to reload
        v16bf pa = load_frag_a(pls, 32, lane);
        for (int j = 0; j < 8; j++) {
            v16bf vb = load_frag_b(Vt + (size_t)(j * 16) * SS + ks, SS, lane);
            accO[j] = wmma_bf16(pa, vb, accO[j]);
        }
    }

    for (int r = 0; r < 8; r++) {
        float invl = 1.f / li[r];
        int row = qs + m_off + r;
        __bf16* dst = aob + ((size_t)(b * SS + row)) * DD + h * DH;
        for (int j = 0; j < 8; j++)
            dst[j * 16 + nidx] = (__bf16)(accO[j][r] * invl);
    }
}

// ---------------- SwiGLU: g <- silu(g) * u (bf16 in-place on g) ---------------
__global__ __launch_bounds__(256) void k_swiglu(__bf16* __restrict__ g,
                                                const __bf16* __restrict__ u,
                                                long long n) {
    long long i = (long long)blockIdx.x * 256 + threadIdx.x;
    if (i >= n) return;
    float gv = (float)g[i], uv = (float)u[i];
    float sv = gv / (1.f + __expf(-gv));
    g[i] = (__bf16)(sv * uv);
}

// ---------------- host-side orchestration ----------------
extern "C" void kernel_launch(void* const* d_in, const int* in_sizes, int n_in,
                              void* d_out, int out_size, void* d_ws, size_t ws_size,
                              hipStream_t stream) {
    (void)in_sizes; (void)n_in; (void)out_size; (void)ws_size;
    const int*   ids   = (const int*)  d_in[0];
    const float* embed = (const float*)d_in[1];
    const float* Wq    = (const float*)d_in[2];
    const float* bq    = (const float*)d_in[3];
    const float* Wk    = (const float*)d_in[4];
    const float* bk    = (const float*)d_in[5];
    const float* Wv    = (const float*)d_in[6];
    const float* bv    = (const float*)d_in[7];
    const float* Wo    = (const float*)d_in[8];
    const float* bo    = (const float*)d_in[9];
    const float* ln1   = (const float*)d_in[10];
    const float* ln2   = (const float*)d_in[11];
    const float* Wg    = (const float*)d_in[12];
    const float* Wu    = (const float*)d_in[13];
    const float* Wd    = (const float*)d_in[14];
    const float* normf = (const float*)d_in[15];

    char* ws = (char*)d_ws;
    size_t off = 0;
    auto alloc = [&](size_t bytes) -> void* {
        void* p = ws + off;
        off += (bytes + 255) & ~(size_t)255;
        return p;
    };
    float*  h   = (float*)  alloc((size_t)MM * DD * 4);
    __bf16* xb  = (__bf16*) alloc((size_t)MM * DD * 2);
    float*  qf  = (float*)  alloc((size_t)MM * DD * 4);
    float*  kf  = (float*)  alloc((size_t)MM * KVD * 4);
    float*  vf  = (float*)  alloc((size_t)MM * KVD * 4);
    __bf16* qb  = (__bf16*) alloc((size_t)MM * DD * 2);
    __bf16* kbf = (__bf16*) alloc((size_t)MM * KVD * 2);
    __bf16* vt  = (__bf16*) alloc((size_t)MM * KVD * 2);
    __bf16* aob = (__bf16*) alloc((size_t)MM * DD * 2);
    __bf16* gb  = (__bf16*) alloc((size_t)MM * II * 2);
    __bf16* ub  = (__bf16*) alloc((size_t)MM * II * 2);
    __bf16* WqT = (__bf16*) alloc((size_t)DD * DD * 2);
    __bf16* WkT = (__bf16*) alloc((size_t)KVD * DD * 2);
    __bf16* WvT = (__bf16*) alloc((size_t)KVD * DD * 2);
    __bf16* WoT = (__bf16*) alloc((size_t)DD * DD * 2);
    __bf16* WgT = (__bf16*) alloc((size_t)II * DD * 2);
    __bf16* WuT = (__bf16*) alloc((size_t)II * DD * 2);
    __bf16* WdT = (__bf16*) alloc((size_t)DD * II * 2);

    // embedding
    k_embed<<<MM, 256, 0, stream>>>(ids, embed, h);

    for (int l = 0; l < LL; l++) {
        const float* wq = Wq + (size_t)l * DD * DD;
        const float* wk = Wk + (size_t)l * DD * KVD;
        const float* wv = Wv + (size_t)l * DD * KVD;
        const float* wo = Wo + (size_t)l * DD * DD;
        const float* wg = Wg + (size_t)l * DD * II;
        const float* wu = Wu + (size_t)l * DD * II;
        const float* wd = Wd + (size_t)l * II * DD;

        // weight conversions (f32 [K][N] -> bf16 [N][K])
        k_cvt_wT<<<dim3(DD/32,  DD/32), 256, 0, stream>>>(wq, WqT, DD, DD);
        k_cvt_wT<<<dim3(KVD/32, DD/32), 256, 0, stream>>>(wk, WkT, DD, KVD);
        k_cvt_wT<<<dim3(KVD/32, DD/32), 256, 0, stream>>>(wv, WvT, DD, KVD);
        k_cvt_wT<<<dim3(DD/32,  DD/32), 256, 0, stream>>>(wo, WoT, DD, DD);
        k_cvt_wT<<<dim3(II/32,  DD/32), 256, 0, stream>>>(wg, WgT, DD, II);
        k_cvt_wT<<<dim3(II/32,  DD/32), 256, 0, stream>>>(wu, WuT, DD, II);
        k_cvt_wT<<<dim3(DD/32,  II/32), 256, 0, stream>>>(wd, WdT, II, DD);

        // attention block
        k_rmsnorm<<<MM, 256, 0, stream>>>(h, ln1 + (size_t)l * DD, xb, nullptr);
        k_gemm<<<dim3(DD/256,  MM/32), 256, 0, stream>>>(xb, WqT, bq + (size_t)l * DD,
                                                         qf, nullptr, MM, DD, DD, 1);
        k_gemm<<<dim3(KVD/256, MM/32), 256, 0, stream>>>(xb, WkT, bk + (size_t)l * KVD,
                                                         kf, nullptr, MM, KVD, DD, 1);
        k_gemm<<<dim3(KVD/256, MM/32), 256, 0, stream>>>(xb, WvT, bv + (size_t)l * KVD,
                                                         vf, nullptr, MM, KVD, DD, 1);
        k_rope<<<dim3(MM, HH + KVH), 64, 0, stream>>>(qf, kf, qb, kbf);
        k_vT<<<MM, 256, 0, stream>>>(vf, vt);
        k_attn<<<dim3(SS/16, HH, BB), 32, 0, stream>>>(qb, kbf, vt, aob);
        k_gemm<<<dim3(DD/256, MM/32), 256, 0, stream>>>(aob, WoT, bo + (size_t)l * DD,
                                                        h, nullptr, MM, DD, DD, 1 | 2);

        // MLP block
        k_rmsnorm<<<MM, 256, 0, stream>>>(h, ln2 + (size_t)l * DD, xb, nullptr);
        k_gemm<<<dim3(II/256, MM/32), 256, 0, stream>>>(xb, WgT, nullptr,
                                                        nullptr, gb, MM, II, DD, 4);
        k_gemm<<<dim3(II/256, MM/32), 256, 0, stream>>>(xb, WuT, nullptr,
                                                        nullptr, ub, MM, II, DD, 4);
        long long nmlp = (long long)MM * II;
        k_swiglu<<<(unsigned)((nmlp + 255) / 256), 256, 0, stream>>>(gb, ub, nmlp);
        k_gemm<<<dim3(DD/256, MM/32), 256, 0, stream>>>(gb, WdT, nullptr,
                                                        h, nullptr, MM, DD, II, 2);
    }

    // final norm -> fp32 output
    k_rmsnorm<<<MM, 256, 0, stream>>>(h, normf, nullptr, (float*)d_out);
}
